// BertSelfAttention_58969900974659
// MI455X (gfx1250) — matrix operand order repaired and checked by reading, not compiled, not run on hardware
//
#include <hip/hip_runtime.h>
#include <hip/hip_bf16.h>

// ---------------------------------------------------------------------------
// BERT self-attention for MI455X (gfx1250, wave32, WMMA + TDM).
// B=4, S=2048, D=1024, H=16, HD=64.
// Pipeline:
//   cvt_f32_bf16 (X, Wq, Wk, Wv -> bf16, one pass)
//   qkv_gemm     (bf16 WMMA, f32 acc; double-buffered K-loop)
//   rope_kernel  (in-place RoPE on bf16 Q,K)
//   flash_attn   (K/V tiles staged to LDS via tensor_load_to_lds (TDM),
//                 double-buffered; QK^T + online softmax + PV on WMMA)
// ---------------------------------------------------------------------------

typedef __attribute__((ext_vector_type(16))) __bf16 v16bf;
typedef __attribute__((ext_vector_type(8)))  float  v8f;
typedef __attribute__((ext_vector_type(4)))  unsigned int v4u;
typedef __attribute__((ext_vector_type(8)))  int    v8i;
typedef __attribute__((ext_vector_type(4)))  int    v4i;

#define BB 4
#define SS 2048
#define DD 1024
#define HH 16
#define HDD 64

#define WMMA_BF16(A, B, C) \
  __builtin_amdgcn_wmma_f32_16x16x32_bf16(false, (A), false, (B), (short)0, (C), false, false)

// 16 contiguous bf16 (32B) -> B-matrix fragment block (lane-contiguous K).
__device__ __forceinline__ v16bf load16bf(const __bf16* p) {
  union { uint4 u[2]; v16bf v; } t;
  t.u[0] = *(const uint4*)(p);
  t.u[1] = *(const uint4*)(p + 8);
  return t.v;
}
// A-matrix per-lane pattern: 8 bf16 at +0, 8 bf16 at +16 elements.
__device__ __forceinline__ v16bf loadAfrag(const __bf16* p) {
  union { uint4 u[2]; v16bf v; } t;
  t.u[0] = *(const uint4*)(p);
  t.u[1] = *(const uint4*)(p + 16);
  return t.v;
}

// ---------------------------------------------------------------------------
// TDM: issue a 2D tile DMA (global -> LDS), bf16 elements (data_size = 2B).
// D# built per CDNA5 ISA 8.3/8.4: group0 = {count, lds_addr, global_addr,
// type=2}; group1 = {data_size, tensor_dim0/1, tile_dim0/1, dim0_stride}.
// Groups 2/3 zero (2D tile). Tracked with TENSORcnt.
// This toolchain's builtin is the 6-arg form:
//   (uint32x4 g0, int32x8 g1, int32x4, int32x4, int32x8, i32 cpol)
// ---------------------------------------------------------------------------
__device__ __forceinline__ void tdm_load_2d(
    unsigned lds_off, const void* gaddr,
    unsigned tdim0, unsigned tdim1,
    unsigned tile0, unsigned tile1, unsigned stride0)
{
  const unsigned long long ga = (unsigned long long)gaddr;
  v4u g0;
  g0[0] = 1u;                                   // count=1 (valid), user mode
  g0[1] = lds_off;                              // lds_addr (bytes)
  g0[2] = (unsigned)ga;                         // global_addr[31:0]
  g0[3] = (unsigned)(ga >> 32) | (2u << 30);    // global_addr[56:32] | type=2
  v8i g1;
  g1[0] = (int)(1u << 16);                      // data_size = 1 -> 2 bytes
  g1[1] = (int)((tdim0 & 0xFFFFu) << 16);       // tensor_dim0[15:0] @ bits 63:48
  g1[2] = (int)((tdim0 >> 16) | ((tdim1 & 0xFFFFu) << 16));
  g1[3] = (int)((tdim1 >> 16) | (tile0 << 16)); // tile_dim0 @ bits 127:112
  g1[4] = (int)tile1;                           // tile_dim1; tile_dim2 = 0
  g1[5] = (int)stride0;                         // tensor_dim0_stride[31:0]
  g1[6] = 0;
  g1[7] = 0;
  v4i z4 = {};
  v8i z8 = {};
  __builtin_amdgcn_tensor_load_to_lds(g0, g1, z4, z4, z8, 0);
}

// Low 32 bits of a generic pointer to __shared__ = LDS byte offset
// (aperture scheme: offset is the low 32 address bits).
__device__ __forceinline__ unsigned lds_offset(const void* p) {
  return (unsigned)(unsigned long long)p;
}

// ---------------------------------------------------------------------------
// Kernel 0: f32 -> bf16 bulk conversion (8 elements / thread, coalesced).
// ---------------------------------------------------------------------------
__global__ __launch_bounds__(256) void cvt_f32_bf16(
    const float* __restrict__ src, __bf16* __restrict__ dst, int n8)
{
  const int i = blockIdx.x * blockDim.x + threadIdx.x;
  if (i >= n8) return;
  const float4* s = (const float4*)src + (size_t)i * 2;
  const float4 a = s[0], b = s[1];
  union { __bf16 h[8]; uint4 u; } t;
  t.h[0] = (__bf16)a.x; t.h[1] = (__bf16)a.y;
  t.h[2] = (__bf16)a.z; t.h[3] = (__bf16)a.w;
  t.h[4] = (__bf16)b.x; t.h[5] = (__bf16)b.y;
  t.h[6] = (__bf16)b.z; t.h[7] = (__bf16)b.w;
  ((uint4*)dst)[i] = t.u;
}

// ---------------------------------------------------------------------------
// Kernel 1: QKV projection GEMM (bf16), software-pipelined / double-buffered.
//   y[m,e] = sum_d X[m,d] * W[e,d]   (x @ W.T; both operands K-contiguous)
// Wave tile: 64(M) x 16(N) = 4 accumulators, B-fragment reused 4x.
// ---------------------------------------------------------------------------
__global__ __launch_bounds__(256) void qkv_gemm(
    const __bf16* __restrict__ Xb, const __bf16* __restrict__ Wb,
    __bf16* __restrict__ Qb, __bf16* __restrict__ Kb, __bf16* __restrict__ Vt)
{
  const int lane = threadIdx.x & 31;
  const int wv   = threadIdx.x >> 5;          // 8 waves / block
  const int mat  = blockIdx.z;                // 0=Q 1=K 2=V
  const __bf16* W = Wb + (size_t)mat * DD * DD;
  const int mbase = blockIdx.x * 64;                       // M = 8192
  const int nbase = (blockIdx.y * 8 + wv) * 16;            // N = 1024
  const int lr    = lane & 15;
  const bool hi   = lane >= 16;

  const __bf16* wp  = W  + (size_t)(nbase + lr) * DD + (hi ? 16 : 0);
  const __bf16* ap0 = Xb + (size_t)(mbase + lr) * DD + (hi ? 16 : 0);
  const __bf16* ap1 = ap0 + (size_t)16 * DD;
  const __bf16* ap2 = ap0 + (size_t)32 * DD;
  const __bf16* ap3 = ap0 + (size_t)48 * DD;

  v8f acc[4] = {};

  v16bf bf0 = load16bf(wp);
  v16bf a00 = load16bf(ap0), a01 = load16bf(ap1),
        a02 = load16bf(ap2), a03 = load16bf(ap3);

  for (int kb = 0; kb < DD; kb += 64) {
    v16bf bf1 = load16bf(wp + kb + 32);
    v16bf a10 = load16bf(ap0 + kb + 32), a11 = load16bf(ap1 + kb + 32),
          a12 = load16bf(ap2 + kb + 32), a13 = load16bf(ap3 + kb + 32);

    acc[0] = WMMA_BF16(a00, bf0, acc[0]);
    acc[1] = WMMA_BF16(a01, bf0, acc[1]);
    acc[2] = WMMA_BF16(a02, bf0, acc[2]);
    acc[3] = WMMA_BF16(a03, bf0, acc[3]);

    if (kb + 64 < DD) {
      bf0 = load16bf(wp + kb + 64);
      a00 = load16bf(ap0 + kb + 64); a01 = load16bf(ap1 + kb + 64);
      a02 = load16bf(ap2 + kb + 64); a03 = load16bf(ap3 + kb + 64);
    }

    acc[0] = WMMA_BF16(a10, bf1, acc[0]);
    acc[1] = WMMA_BF16(a11, bf1, acc[1]);
    acc[2] = WMMA_BF16(a12, bf1, acc[2]);
    acc[3] = WMMA_BF16(a13, bf1, acc[3]);
  }

  const int e = nbase + lr, h = e >> 6, hd = e & 63;
  #pragma unroll
  for (int sub = 0; sub < 4; ++sub) {
    #pragma unroll
    for (int r = 0; r < 8; ++r) {
      const int gm = mbase + sub * 16 + r + (hi ? 8 : 0);
      const int b  = gm >> 11;
      const int s  = gm & 2047;
      if (mat == 2) {
        Vt[(((size_t)b * HH + h) * HDD + hd) * SS + s] = (__bf16)acc[sub][r];
      } else {
        __bf16* dst = (mat == 0) ? Qb : Kb;
        dst[(((size_t)b * HH + h) * SS + s) * HDD + hd] = (__bf16)acc[sub][r];
      }
    }
  }
}

// ---------------------------------------------------------------------------
// Kernel 2: RoPE in-place on bf16 Q and K ([B,H,S,HD] layout).
// ---------------------------------------------------------------------------
__global__ __launch_bounds__(256) void rope_kernel(
    __bf16* __restrict__ Qb, __bf16* __restrict__ Kb,
    const float* __restrict__ cosb, const float* __restrict__ sinb)
{
  const int idx = blockIdx.x * blockDim.x + threadIdx.x;
  const int j   = idx & 31;
  const int s   = (idx >> 5) & 2047;
  const int bh  = (idx >> 16) & 63;
  __bf16* P = (idx >> 22) ? Kb : Qb;
  const size_t base = ((size_t)bh * SS + s) * HDD;
  const float x1 = (float)P[base + j];
  const float x2 = (float)P[base + j + 32];
  const float c  = cosb[s * HDD + j];
  const float sn = sinb[s * HDD + j];
  P[base + j]      = (__bf16)(x1 * c - x2 * sn);
  P[base + j + 32] = (__bf16)(x2 * c + x1 * sn);
}

// ---------------------------------------------------------------------------
// Kernel 3: flash attention with TDM-staged K/V tiles.
// grid = (S/128, B*H), 8 waves/block; all waves share one (b,h) K/V stream.
// Per 32-key step: wave 0 issues TDM loads of the *next* K tile (32x64 bf16)
// and V tile (64x32 bf16) into the other LDS buffer, all waves compute from
// the current buffer (ds_load fragments), then s_wait_tensorcnt + barrier.
// ---------------------------------------------------------------------------
__global__ __launch_bounds__(256) void flash_attn(
    const __bf16* __restrict__ Qb, const __bf16* __restrict__ Kb,
    const __bf16* __restrict__ Vt, float* __restrict__ out)
{
  __shared__ __bf16 Kt[2][32][64];     // 2 x 4 KB  (keys x head-dim)
  __shared__ __bf16 Vtile[2][64][32];  // 2 x 4 KB  (head-dim x keys)
  __shared__ __bf16 Pbuf[8][16][32];   // 8 KB per-wave P transpose staging

  const int lane  = threadIdx.x & 31;
  const int wv    = threadIdx.x >> 5;
  const int lr    = lane & 15;
  const bool hiH  = lane >= 16;
  const int bh    = blockIdx.y;
  const int qbase = blockIdx.x * 128 + wv * 16;

  const __bf16* Qp = Qb + (size_t)bh * SS * HDD;
  const __bf16* Kp = Kb + (size_t)bh * SS * HDD;
  const __bf16* Vp = Vt + (size_t)bh * HDD * SS;

  // Q A-fragments (16x32 each, two K-steps covering HD=64) held all loop.
  const v16bf qa0 = load16bf(Qp + (size_t)(qbase + lr) * HDD + (hiH ? 16 : 0));
  const v16bf qa1 = load16bf(Qp + (size_t)(qbase + lr) * HDD + 32 + (hiH ? 16 : 0));

  float mrow[8], lrow[8];
  v8f acc[4] = {};
  #pragma unroll
  for (int r = 0; r < 8; ++r) { mrow[r] = -1e30f; lrow[r] = 0.f; }

  // TDM: stage K/V tiles for key block kb into LDS buffer `buf`.
  auto stage = [&](int kb, int buf) {
    // K tile: tensor 64(x) x 2048(y), tile 64 x 32, row stride 64.
    tdm_load_2d(lds_offset(&Kt[buf][0][0]), Kp + (size_t)kb * HDD,
                HDD, SS, HDD, 32, HDD);
    // V tile: tensor 2048(x) x 64(y), tile 32 x 64, row stride 2048.
    tdm_load_2d(lds_offset(&Vtile[buf][0][0]), Vp + kb,
                SS, HDD, 32, HDD, SS);
  };

  // One 32-key step out of LDS buffer `buf`.
  auto step = [&](int buf) {
    // K B-fragments from LDS (2 key-tiles x 2 d-halves).
    const v16bf f00 = load16bf(&Kt[buf][lr][hiH ? 16 : 0]);
    const v16bf f01 = load16bf(&Kt[buf][lr][32 + (hiH ? 16 : 0)]);
    const v16bf f10 = load16bf(&Kt[buf][16 + lr][hiH ? 16 : 0]);
    const v16bf f11 = load16bf(&Kt[buf][16 + lr][32 + (hiH ? 16 : 0)]);
    // V B-fragments from LDS (4 d-tiles x 32 keys).
    const v16bf vf0 = load16bf(&Vtile[buf][lr][hiH ? 16 : 0]);
    const v16bf vf1 = load16bf(&Vtile[buf][16 + lr][hiH ? 16 : 0]);
    const v16bf vf2 = load16bf(&Vtile[buf][32 + lr][hiH ? 16 : 0]);
    const v16bf vf3 = load16bf(&Vtile[buf][48 + lr][hiH ? 16 : 0]);

    // Scores for two 16-key tiles (K-dim = HD = 64 -> 2 WMMA each).
    v8f z0 = {}, z1 = {};
    z0 = WMMA_BF16(qa0, f00, z0);
    z1 = WMMA_BF16(qa0, f10, z1);
    v8f sc0 = WMMA_BF16(qa1, f01, z0);
    v8f sc1 = WMMA_BF16(qa1, f11, z1);
    #pragma unroll
    for (int r = 0; r < 8; ++r) { sc0[r] *= 0.125f; sc1[r] *= 0.125f; }

    // Online softmax: row reductions across the 16 lanes of each half.
    float corr[8];
    #pragma unroll
    for (int r = 0; r < 8; ++r) {
      float v = fmaxf(sc0[r], sc1[r]);
      v = fmaxf(v, __shfl_xor(v, 1, 32));
      v = fmaxf(v, __shfl_xor(v, 2, 32));
      v = fmaxf(v, __shfl_xor(v, 4, 32));
      v = fmaxf(v, __shfl_xor(v, 8, 32));
      const float mnew = fmaxf(mrow[r], v);
      corr[r] = exp2f((mrow[r] - mnew) * 1.44269504f);
      mrow[r] = mnew;
    }
    #pragma unroll
    for (int r = 0; r < 8; ++r) {
      const float p0 = exp2f((sc0[r] - mrow[r]) * 1.44269504f);
      const float p1 = exp2f((sc1[r] - mrow[r]) * 1.44269504f);
      float rs = p0 + p1;
      rs += __shfl_xor(rs, 1, 32);
      rs += __shfl_xor(rs, 2, 32);
      rs += __shfl_xor(rs, 4, 32);
      rs += __shfl_xor(rs, 8, 32);
      lrow[r] = lrow[r] * corr[r] + rs;
      const int row = r + (hiH ? 8 : 0);
      Pbuf[wv][row][lr]      = (__bf16)p0;   // C-layout -> row-major LDS
      Pbuf[wv][row][lr + 16] = (__bf16)p1;
    }
    asm volatile("s_wait_dscnt 0" ::: "memory");  // P writes visible to reads

    const v16bf pa = load16bf(&Pbuf[wv][lr][hiH ? 16 : 0]);

    #pragma unroll
    for (int r = 0; r < 8; ++r) {
      acc[0][r] *= corr[r]; acc[1][r] *= corr[r];
      acc[2][r] *= corr[r]; acc[3][r] *= corr[r];
    }
    acc[0] = WMMA_BF16(pa, vf0, acc[0]);
    acc[1] = WMMA_BF16(pa, vf1, acc[1]);
    acc[2] = WMMA_BF16(pa, vf2, acc[2]);
    acc[3] = WMMA_BF16(pa, vf3, acc[3]);
  };

  // Prologue: fill buffer 0, wait TDM, sync.
  if (wv == 0) {
    stage(0, 0);
    __builtin_amdgcn_s_wait_tensorcnt(0);
  }
  __syncthreads();

  for (int kb = 0; kb < SS; kb += 32) {
    const int buf = (kb >> 5) & 1;
    if (wv == 0 && kb + 32 < SS)
      stage(kb + 32, buf ^ 1);     // TDM fills next buffer during compute
    step(buf);
    if (wv == 0)
      __builtin_amdgcn_s_wait_tensorcnt(0);
    __syncthreads();
  }

  // Epilogue: out[b, s, h*64+d] = acc / l.
  const int b = bh >> 4, h = bh & 15;
  #pragma unroll
  for (int dt = 0; dt < 4; ++dt) {
    #pragma unroll
    for (int r = 0; r < 8; ++r) {
      const int s = qbase + r + (hiH ? 8 : 0);
      const int d = h * HDD + dt * 16 + lr;
      out[((size_t)b * SS + s) * DD + d] = acc[dt][r] / lrow[r];
    }
  }
}

// ---------------------------------------------------------------------------
extern "C" void kernel_launch(void* const* d_in, const int* in_sizes, int n_in,
                              void* d_out, int out_size, void* d_ws, size_t ws_size,
                              hipStream_t stream) {
  (void)in_sizes; (void)n_in; (void)out_size; (void)ws_size;
  const float* X    = (const float*)d_in[0];
  // d_in[1] = attention_mask (ignored by reference)
  const float* cosb = (const float*)d_in[2];
  const float* sinb = (const float*)d_in[3];
  const float* Wq   = (const float*)d_in[4];
  const float* Wk   = (const float*)d_in[5];
  const float* Wv   = (const float*)d_in[6];

  const size_t elems  = (size_t)BB * SS * DD;  // 8,388,608
  const size_t welems = (size_t)DD * DD;       // 1,048,576
  __bf16* Qb = (__bf16*)d_ws;                  // 16 MB
  __bf16* Kb = Qb + elems;                     // 16 MB
  __bf16* Vt = Kb + elems;                     // 16 MB (transposed [B,H,HD,S])
  __bf16* Xb = Vt + elems;                     // 16 MB
  __bf16* Wb = Xb + elems;                     //  6 MB (Wq|Wk|Wv)

  // Phase 0: one-shot f32 -> bf16 conversion (bandwidth-bound, L2 resident).
  const int x8 = (int)(elems / 8);
  cvt_f32_bf16<<<(x8 + 255) / 256, 256, 0, stream>>>(X, Xb, x8);
  const int w8 = (int)(welems / 8);
  cvt_f32_bf16<<<(w8 + 255) / 256, 256, 0, stream>>>(Wq, Wb,              w8);
  cvt_f32_bf16<<<(w8 + 255) / 256, 256, 0, stream>>>(Wk, Wb + welems,     w8);
  cvt_f32_bf16<<<(w8 + 255) / 256, 256, 0, stream>>>(Wv, Wb + 2 * welems, w8);

  // QKV projections: M=8192 (x-dim, 64 rows/block), N=1024 (y*8 waves), z={Q,K,V}
  dim3 g1(128, 8, 3);
  qkv_gemm<<<g1, 256, 0, stream>>>(Xb, Wb, Qb, Kb, Vt);

  // RoPE on Q and K: 2 * 64 * 2048 * 32 pair-threads.
  const int rope_threads = 2 * BB * HH * SS * 32;
  rope_kernel<<<rope_threads / 256, 256, 0, stream>>>(Qb, Kb, cosb, sinb);

  // Flash attention: 128 queries/block, one (b,h) per y.
  dim3 g2(SS / 128, BB * HH);
  flash_attn<<<g2, 256, 0, stream>>>(Qb, Kb, Vt, (float*)d_out);
}
